// AttentionHead_50594714747512
// MI455X (gfx1250) — matrix operand order repaired and checked by reading, not compiled, not run on hardware
//
#include <hip/hip_runtime.h>
#include <hip/hip_bf16.h>

#define B_DIM 4
#define T_LEN 4096
#define D_MODEL 1024
#define D_HEAD 64
#define QTILES_PER_B (T_LEN / 16)   // 256

static constexpr float ATTN_SCALE = 0.125f;  // 64^-0.5

typedef __attribute__((ext_vector_type(16))) __bf16 v16bf;
typedef __attribute__((ext_vector_type(8)))  float  v8f;
typedef unsigned int u32x4 __attribute__((ext_vector_type(4)));
typedef int          i32x8 __attribute__((ext_vector_type(8)));
typedef int          i32x4 __attribute__((ext_vector_type(4)));

#if defined(__has_builtin)
#if __has_builtin(__builtin_amdgcn_tensor_load_to_lds)
#define HAVE_TDM 1
#endif
#endif

union Frag {
    v16bf v;
    uint4 q2[2];
};

__device__ __forceinline__ unsigned short f2bf(float f) {
    unsigned int u = __float_as_uint(f);
    u += 0x7fffu + ((u >> 16) & 1u);   // round-to-nearest-even
    return (unsigned short)(u >> 16);
}

__device__ __forceinline__ v8f wmma_bf16(const Frag& a, const Frag& b, v8f c) {
    return __builtin_amdgcn_wmma_f32_16x16x32_bf16(
        /*neg_a=*/false, a.v, /*neg_b=*/false, b.v,
        /*c_mod=*/(short)0, c, /*reuse_a=*/false, /*reuse_b=*/false);
}

__device__ __forceinline__ float rmax16(float v) {
#pragma unroll
    for (int m = 1; m < 16; m <<= 1) v = fmaxf(v, __shfl_xor(v, m, 32));
    return v;
}
__device__ __forceinline__ float rsum16(float v) {
#pragma unroll
    for (int m = 1; m < 16; m <<= 1) v += __shfl_xor(v, m, 32);
    return v;
}

// ---------------------------------------------------------------------------
// TDM: stage a [192 rows x 32 bf16] tile (row stride 1024 bf16) from global
// into LDS at lds_off, padding each 64B row with 16B -> 80B LDS row stride.
// D# fields per CDNA5 ISA.
// ---------------------------------------------------------------------------
#ifdef HAVE_TDM
__device__ __forceinline__ void tdm_stage_w(const unsigned short* gsrc,
                                            unsigned int lds_off) {
    const unsigned long long ga = (unsigned long long)(uintptr_t)gsrc;
    u32x4 g0;
    g0[0] = 1u;                                        // count=1 (valid)
    g0[1] = lds_off;                                   // lds_addr
    g0[2] = (unsigned int)(ga & 0xFFFFFFFFu);          // global_addr[31:0]
    g0[3] = (unsigned int)((ga >> 32) & 0x01FFFFFFu)   // global_addr[56:32]
            | (2u << 30);                              // type=2 ("image")
    i32x8 g1;
    g1[0] = (int)((1u << 16)    // data_size = 2 bytes
                | (1u << 20)    // pad_enable
                | (3u << 22)    // pad_interval: 16 DWORDs (64B) before pad
                | (3u << 25));  // pad_amount: 4 DWORDs (16B)
    g1[1] = (int)(1024u << 16); // tensor_dim0 = 1024 (bits 79:48, low part)
    g1[2] = (int)(192u << 16);  // tensor_dim1 = 192  (bits 111:80)
    g1[3] = (int)(32u << 16);   // tile_dim0 = 32     (bits 127:112)
    g1[4] = 192;                // tile_dim1 = 192    (bits 143:128)
    g1[5] = 1024;               // tensor_dim0_stride = 1024 (bits 207:160)
    g1[6] = 0;
    g1[7] = 0;
    const i32x4 z4 = {0, 0, 0, 0};
#if __has_include(<hip/amd_detail/amd_gfx1250_TDM.h>)
    const i32x8 z8 = {0, 0, 0, 0, 0, 0, 0, 0};
    __builtin_amdgcn_tensor_load_to_lds(g0, g1, z4, z4, z8, 0);
#else
    __builtin_amdgcn_tensor_load_to_lds(g0, g1, z4, z4, 0);
#endif
}
#endif

// ---------------------------------------------------------------------------
// Kernel 1a: fp32 -> bf16 convert of x (vectorized x4)
// ---------------------------------------------------------------------------
__global__ void ah_convert_x(const float* __restrict__ x,
                             unsigned short* __restrict__ xb, int n4) {
    int i = blockIdx.x * blockDim.x + threadIdx.x;
    if (i < n4) {
        float4 f = reinterpret_cast<const float4*>(x)[i];
        ushort4 o;
        o.x = f2bf(f.x); o.y = f2bf(f.y); o.z = f2bf(f.z); o.w = f2bf(f.w);
        reinterpret_cast<ushort4*>(xb)[i] = o;
    }
}

// ---------------------------------------------------------------------------
// Kernel 1b: Wq/Wk/Wv [1024,64] fp32 -> Wt [3][64][1024] bf16 (transposed)
// ---------------------------------------------------------------------------
__global__ void ah_convert_w(const float* __restrict__ Wq,
                             const float* __restrict__ Wk,
                             const float* __restrict__ Wv,
                             unsigned short* __restrict__ wt) {
    int i = blockIdx.x * blockDim.x + threadIdx.x;
    if (i < 3 * D_HEAD * D_MODEL) {
        int m = i >> 16;          // / (64*1024)
        int r = i & 65535;
        int n = r >> 10;          // / 1024
        int k = r & 1023;
        const float* W = (m == 0) ? Wq : ((m == 1) ? Wk : Wv);
        wt[i] = f2bf(W[k * D_HEAD + n]);
    }
}

// ---------------------------------------------------------------------------
// Kernel 2: QKV projection. 4 waves/block, each wave one 16-row time tile.
// Wt K-chunks staged into LDS by the Tensor Data Mover (double-buffered).
//   q,k -> bf16 [B,T,64];  v -> bf16 transposed [B,64,T]
// ---------------------------------------------------------------------------
#define WROW 40   // 32 data ushorts + 8 pad -> 80B row stride (20 banks, 16B aligned)

__global__ __launch_bounds__(128, 1)
void ah_qkv(const unsigned short* __restrict__ xb,
            const unsigned short* __restrict__ wt,
            const float* __restrict__ bq, const float* __restrict__ bk,
            const float* __restrict__ bv,
            unsigned short* __restrict__ qout,
            unsigned short* __restrict__ kout,
            unsigned short* __restrict__ vtout) {
    __shared__ __align__(16) unsigned short sW[2][192 * WROW];

    const int tid  = threadIdx.x;
    const int wave = tid >> 5;
    const int lane = tid & 31;
    const int half = lane >> 4;
    const int l16  = lane & 15;
    const int tile = blockIdx.x * 4 + wave;
    const int b    = tile >> 8;           // / QTILES_PER_B
    const int t0   = (tile & 255) << 4;
    const int abase = half * 16;
    const int bbase = half * 16;

    v8f acc[3][4];
    const v8f vzero = {0.f, 0.f, 0.f, 0.f, 0.f, 0.f, 0.f, 0.f};
#pragma unroll
    for (int m = 0; m < 3; ++m)
#pragma unroll
        for (int n = 0; n < 4; ++n) acc[m][n] = vzero;

    const unsigned short* xrow =
        xb + ((size_t)(b * T_LEN) + t0 + l16) * D_MODEL;

#ifdef HAVE_TDM
    const unsigned int lds0 = (unsigned int)(uintptr_t)&sW[0][0];
    const unsigned int lds1 = (unsigned int)(uintptr_t)&sW[1][0];
    if (wave == 0) {
        tdm_stage_w(wt, lds0);
        __builtin_amdgcn_s_wait_tensorcnt(0);
    }
#else
    // Fallback: register staging (384 uint4 / 128 threads = 3 per thread)
    uint4 sr[3];
    auto load_stage = [&](int c) {
#pragma unroll
        for (int i = 0; i < 3; ++i) {
            const int it = tid + i * 128;
            const int r  = it >> 1;
            const int h  = it & 1;
            sr[i] = *reinterpret_cast<const uint4*>(
                wt + (size_t)r * D_MODEL + c * 32 + h * 8);
        }
    };
    auto store_stage = [&](int buf) {
#pragma unroll
        for (int i = 0; i < 3; ++i) {
            const int it = tid + i * 128;
            const int r  = it >> 1;
            const int h  = it & 1;
            *reinterpret_cast<uint4*>(&sW[buf][r * WROW + h * 8]) = sr[i];
        }
    };
    load_stage(0);
    store_stage(0);
#endif
    __syncthreads();

    for (int c = 0; c < 32; ++c) {
        const int cur = c & 1;
        const bool more = (c + 1) < 32;

#ifdef HAVE_TDM
        if (more && wave == 0)
            tdm_stage_w(wt + (c + 1) * 32, cur ? lds0 : lds1);  // async DMA
#else
        if (more) load_stage(c + 1);
#endif

        Frag a;
        a.q2[0] = *reinterpret_cast<const uint4*>(xrow + c * 32 + abase);
        a.q2[1] = *reinterpret_cast<const uint4*>(xrow + c * 32 + abase + 8);
        if (more) __builtin_prefetch(xrow + (c + 1) * 32, 0, 3);

        // 12 WMMAs with B-fragments from LDS
#pragma unroll
        for (int m = 0; m < 3; ++m) {
#pragma unroll
            for (int n = 0; n < 4; ++n) {
                const int row = m * 64 + n * 16 + l16;
                Frag w;
                w.q2[0] = *reinterpret_cast<const uint4*>(&sW[cur][row * WROW + bbase]);
                w.q2[1] = *reinterpret_cast<const uint4*>(&sW[cur][row * WROW + bbase + 8]);
                acc[m][n] = wmma_bf16(a, w, acc[m][n]);
            }
        }

#ifdef HAVE_TDM
        if (more && wave == 0) __builtin_amdgcn_s_wait_tensorcnt(0);
#else
        if (more) store_stage(cur ^ 1);
#endif
        __syncthreads();
    }

#pragma unroll
    for (int n = 0; n < 4; ++n) {
        const int d = n * 16 + l16;
        const float bqa = bq[d], bka = bk[d], bva = bv[d];
#pragma unroll
        for (int j = 0; j < 8; ++j) {
            const int t = t0 + j + half * 8;
            const size_t base = ((size_t)(b * T_LEN) + t) * D_HEAD + d;
            qout[base] = f2bf(acc[0][n][j] + bqa);
            kout[base] = f2bf(acc[1][n][j] + bka);
            vtout[((size_t)b * D_HEAD + d) * T_LEN + t] = f2bf(acc[2][n][j] + bva);
        }
    }
}

// ---------------------------------------------------------------------------
// Kernel 3: flash attention, intra-block split-K.
// 4 waves/block share one 16-query tile; wave w handles key blocks
// kb = w, w+4, ... (32 keys each) with private online-softmax state,
// then partials are merged through LDS.
// ---------------------------------------------------------------------------
__global__ __launch_bounds__(128, 1)
void ah_attn(const unsigned short* __restrict__ qb,
             const unsigned short* __restrict__ kbuf,
             const unsigned short* __restrict__ vt,
             float* __restrict__ out) {
    __shared__ __align__(16) unsigned short sP[4][16 * 32];  // per-wave P tile
    __shared__ float sO[4][16][64];
    __shared__ float sM[4][16];
    __shared__ float sL[4][16];

    const int tid  = threadIdx.x;
    const int wave = tid >> 5;
    const int lane = tid & 31;
    const int half = lane >> 4;
    const int l16  = lane & 15;
    const int b    = blockIdx.x >> 8;
    const int q0   = (blockIdx.x & 255) << 4;
    const int abase = half * 16;
    const int bbase = half * 16;

    // Q A-fragments (d = 0..31 and 32..63) — same for all waves of the block
    Frag qa0, qa1;
    const unsigned short* qrow = qb + ((size_t)(b * T_LEN) + q0 + l16) * D_HEAD;
    qa0.q2[0] = *reinterpret_cast<const uint4*>(qrow + abase);
    qa0.q2[1] = *reinterpret_cast<const uint4*>(qrow + abase + 8);
    qa1.q2[0] = *reinterpret_cast<const uint4*>(qrow + 32 + abase);
    qa1.q2[1] = *reinterpret_cast<const uint4*>(qrow + 32 + abase + 8);

    const v8f vzero = {0.f, 0.f, 0.f, 0.f, 0.f, 0.f, 0.f, 0.f};
    v8f o[4];
#pragma unroll
    for (int n = 0; n < 4; ++n) o[n] = vzero;

    float mrow[8], lrow[8];
#pragma unroll
    for (int j = 0; j < 8; ++j) { mrow[j] = -__builtin_inff(); lrow[j] = 0.f; }

    const int kbmax = (q0 + 15) >> 5;   // inclusive; causal
    for (int kb = wave; kb <= kbmax; kb += 4) {
        const int k0 = kb << 5;
        const unsigned short* vbase =
            vt + ((size_t)(b * D_HEAD) + l16) * T_LEN + k0 + bbase;

        // Prefetch this iteration's V lines; loads happen after the softmax.
#pragma unroll
        for (int n = 0; n < 4; ++n)
            __builtin_prefetch(vbase + (size_t)(n * 16) * T_LEN, 0, 3);

        // ---- S = Q K^T ----
        float sv[2][8];
#pragma unroll
        for (int sub = 0; sub < 2; ++sub) {
            const int ki = k0 + sub * 16 + l16;
            const unsigned short* krow =
                kbuf + ((size_t)(b * T_LEN) + ki) * D_HEAD;
            __builtin_prefetch(krow + 4 * 32 * D_HEAD, 0, 3);  // next kb
            Frag kf0, kf1;
            kf0.q2[0] = *reinterpret_cast<const uint4*>(krow + bbase);
            kf0.q2[1] = *reinterpret_cast<const uint4*>(krow + bbase + 8);
            kf1.q2[0] = *reinterpret_cast<const uint4*>(krow + 32 + bbase);
            kf1.q2[1] = *reinterpret_cast<const uint4*>(krow + 32 + bbase + 8);
            v8f s = vzero;
            s = wmma_bf16(qa0, kf0, s);
            s = wmma_bf16(qa1, kf1, s);
#pragma unroll
            for (int j = 0; j < 8; ++j) {
                const int qi = q0 + j + half * 8;
                const float val = s[j] * ATTN_SCALE;
                sv[sub][j] = (ki <= qi) ? val : -__builtin_inff();
            }
        }

        // ---- online softmax ----
#pragma unroll
        for (int j = 0; j < 8; ++j) {
            const float rm = rmax16(fmaxf(sv[0][j], sv[1][j]));
            const float nm = fmaxf(mrow[j], rm);
            const float alpha = __expf(mrow[j] - nm);
            const float p0 = __expf(sv[0][j] - nm);
            const float p1 = __expf(sv[1][j] - nm);
            lrow[j] = lrow[j] * alpha + rsum16(p0 + p1);
            mrow[j] = nm;
#pragma unroll
            for (int n = 0; n < 4; ++n) o[n][j] *= alpha;
            const int r = j + half * 8;
            sP[wave][r * 32 + l16]      = f2bf(p0);
            sP[wave][r * 32 + 16 + l16] = f2bf(p1);
        }
        // per-wave private LDS tile; DS ops execute in order within a wave —
        // only keep the compiler from reordering the loads above the stores.
        asm volatile("" ::: "memory");

        // ---- O += P V  (V fragments just-in-time; lines prefetched) ----
        Frag pa;
        pa.q2[0] = *reinterpret_cast<const uint4*>(&sP[wave][l16 * 32 + abase]);
        pa.q2[1] = *reinterpret_cast<const uint4*>(&sP[wave][l16 * 32 + abase + 8]);
#pragma unroll
        for (int n = 0; n < 4; ++n) {
            const unsigned short* vrow = vbase + (size_t)(n * 16) * T_LEN;
            Frag vf;
            vf.q2[0] = *reinterpret_cast<const uint4*>(vrow);
            vf.q2[1] = *reinterpret_cast<const uint4*>(vrow + 8);
            o[n] = wmma_bf16(pa, vf, o[n]);
        }
        asm volatile("" ::: "memory");
    }

    // ---- write per-wave partials ----
#pragma unroll
    for (int j = 0; j < 8; ++j) {
        const int r = j + half * 8;
#pragma unroll
        for (int n = 0; n < 4; ++n) sO[wave][r][n * 16 + l16] = o[n][j];
        if (l16 == 0) { sM[wave][r] = mrow[j]; sL[wave][r] = lrow[j]; }
    }
    __syncthreads();

    // ---- merge: 16 rows x 64 cols = 1024 outputs, 8 per thread ----
#pragma unroll
    for (int i = 0; i < 8; ++i) {
        const int e = tid + i * 128;
        const int r = e >> 6;
        const int d = e & 63;
        float M = sM[0][r];
#pragma unroll
        for (int w = 1; w < 4; ++w) M = fmaxf(M, sM[w][r]);
        float L = 0.f, val = 0.f;
#pragma unroll
        for (int w = 0; w < 4; ++w) {
            const float wgt = __expf(sM[w][r] - M);
            L   += wgt * sL[w][r];
            val += wgt * sO[w][r][d];
        }
        out[((size_t)(b * T_LEN) + q0 + r) * D_HEAD + d] = val / L;
    }
}

// ---------------------------------------------------------------------------
extern "C" void kernel_launch(void* const* d_in, const int* in_sizes, int n_in,
                              void* d_out, int out_size, void* d_ws, size_t ws_size,
                              hipStream_t stream) {
    const float* x  = (const float*)d_in[0];
    const float* Wq = (const float*)d_in[1];
    const float* bq = (const float*)d_in[2];
    const float* Wk = (const float*)d_in[3];
    const float* bk = (const float*)d_in[4];
    const float* Wv = (const float*)d_in[5];
    const float* bv = (const float*)d_in[6];
    float* out = (float*)d_out;

    // Workspace layout (bytes):
    //   xb : B*T*1024 bf16 = 33,554,432
    //   wt : 3*64*1024 bf16 =   393,216
    //   qb : B*T*64  bf16 =  2,097,152
    //   kb : B*T*64  bf16 =  2,097,152
    //   vt : B*64*T  bf16 =  2,097,152
    char* ws = (char*)d_ws;
    unsigned short* xb  = (unsigned short*)(ws);
    unsigned short* wt  = (unsigned short*)(ws + 33554432u);
    unsigned short* qbp = (unsigned short*)(ws + 33554432u + 393216u);
    unsigned short* kbp = (unsigned short*)(ws + 33554432u + 393216u + 2097152u);
    unsigned short* vtp = (unsigned short*)(ws + 33554432u + 393216u + 2u * 2097152u);

    const int nx4 = (B_DIM * T_LEN * D_MODEL) / 4;           // 4,194,304
    ah_convert_x<<<(nx4 + 255) / 256, 256, 0, stream>>>(x, xb, nx4);

    const int nw = 3 * D_HEAD * D_MODEL;                     // 196,608
    ah_convert_w<<<(nw + 255) / 256, 256, 0, stream>>>(Wq, Wk, Wv, wt);

    const int ntiles = B_DIM * QTILES_PER_B;                 // 1024 tiles
    ah_qkv<<<ntiles / 4, 128, 0, stream>>>(xb, wt, bq, bk, bv, qbp, kbp, vtp);

    ah_attn<<<ntiles, 128, 0, stream>>>(qbp, kbp, vtp, out);
}
